// LSTM_32134945308961
// MI455X (gfx1250) — compile-verified, matrix operand order checked by reading, not run on hardware
//
#include <hip/hip_runtime.h>
#include <hip/hip_bf16.h>
#include <math.h>

typedef __bf16 bf16_t;
typedef __attribute__((ext_vector_type(16))) __bf16 v16bf;
typedef __attribute__((ext_vector_type(8)))  float  v8f;

static constexpr int TT  = 256;
static constexpr int BB  = 64;
static constexpr int INS = 512;
static constexpr int HH  = 1024;
static constexpr int GG  = 4096;   // 4*H
static constexpr int APAD = 40;    // LDS A-tile row stride (bf16 elems), 80B = 16B aligned

// ---------------------------------------------------------------------------
// Fragment loaders (wave32 WMMA 16x16x32 bf16 layouts)
// A (16x32, MxK): lane<16 -> row M=lane, K in {0..7, 16..23};
//                 lane>=16 -> row M=lane-16, K in {8..15, 24..31}
// B (32x16, KxN): column n = lane&15 (same half-split of K). With row-major
// weights W[n, k], column n is contiguous over k -> two 16B loads per lane.
// ---------------------------------------------------------------------------
__device__ __forceinline__ v16bf g_frag(const bf16_t* __restrict__ base, int stride,
                                        int row0, int k0, int lane) {
  const int half = lane >> 4;
  const bf16_t* p = base + (size_t)(row0 + (lane & 15)) * stride + k0 + half * 8;
  union { uint4 u[2]; v16bf v; } t;
  t.u[0] = *reinterpret_cast<const uint4*>(p);
  t.u[1] = *reinterpret_cast<const uint4*>(p + 16);
  return t.v;
}

__device__ __forceinline__ v16bf lds_frag(const bf16_t* buf, int m0, int lane) {
  const int half = lane >> 4;
  const bf16_t* p = buf + (m0 + (lane & 15)) * APAD + half * 8;
  union { uint4 u[2]; v16bf v; } t;
  t.u[0] = *reinterpret_cast<const uint4*>(p);
  t.u[1] = *reinterpret_cast<const uint4*>(p + 16);
  return t.v;
}

// Stage a 64-row x 32-col bf16 tile into LDS (256 threads, one 16B load each).
__device__ __forceinline__ void stage64x32(bf16_t* dst, const bf16_t* __restrict__ src,
                                           int row_stride, int k0, int tid) {
  const int r = tid >> 2;          // 0..63
  const int c = (tid & 3) * 8;     // 0,8,16,24
  uint4 v = *reinterpret_cast<const uint4*>(src + (size_t)r * row_stride + k0 + c);
  *reinterpret_cast<uint4*>(dst + r * APAD + c) = v;
}

__device__ __forceinline__ float sigm(float x) { return 1.0f / (1.0f + expf(-x)); }

__device__ __forceinline__ v8f wmma_bf16(v16bf a, v16bf b, v8f c) {
  return __builtin_amdgcn_wmma_f32_16x16x32_bf16(false, a, false, b, (short)0, c,
                                                 false, false);
}

// ---------------------------------------------------------------------------
// fp32 -> bf16 conversion
// ---------------------------------------------------------------------------
__global__ void k_cvt_bf16(const float* __restrict__ x, bf16_t* __restrict__ y, int n) {
  int i = blockIdx.x * blockDim.x + threadIdx.x;
  if (i < n) y[i] = (bf16_t)x[i];
}

// ---------------------------------------------------------------------------
// Kernel A: xproj[t,b,g] = inp[t,b,:] . Wxh[g,:] + bxh[g] (+ img4 at t==0)
// M = T*B = 16384 (4 M-tiles / WG), N = 4096 (128 cols / WG), K = 512.
// grid = (GG/128, (T*B)/64), block = 256 (8 waves).
// wave w: M-tile (w&3); 4 accumulators over N = nbase + (w>>2)*16 + g*32.
// ---------------------------------------------------------------------------
__global__ __launch_bounds__(256) void k_xproj(
    const bf16_t* __restrict__ inp_bf, const bf16_t* __restrict__ Wxh_bf,
    const float* __restrict__ bxh, const float* __restrict__ img,
    float* __restrict__ xproj) {
  __shared__ bf16_t abuf[2][64 * APAD];
  const int tid = threadIdx.x, lane = tid & 31, w = tid >> 5;
  const int half = lane >> 4, nc = lane & 15;
  const int mbase = blockIdx.y * 64;
  const int nbase = blockIdx.x * 128 + (w >> 2) * 16;
  const int mloc = (w & 3) * 16;

  v8f acc[4];
#pragma unroll
  for (int g = 0; g < 4; ++g)
#pragma unroll
    for (int j = 0; j < 8; ++j) acc[g][j] = 0.0f;

  const bf16_t* asrc = inp_bf + (size_t)mbase * INS;
  stage64x32(&abuf[0][0], asrc, INS, 0, tid);
  __syncthreads();
  const int KT = INS / 32;
  for (int kt = 0; kt < KT; ++kt) {
    const int buf = kt & 1;
    if (kt + 1 < KT) stage64x32(&abuf[buf ^ 1][0], asrc, INS, (kt + 1) * 32, tid);
    v16bf a = lds_frag(&abuf[buf][0], mloc, lane);
#pragma unroll
    for (int g = 0; g < 4; ++g) {
      v16bf b = g_frag(Wxh_bf, INS, nbase + g * 32, kt * 32, lane);
      acc[g] = wmma_bf16(a, b, acc[g]);
    }
    __syncthreads();
  }

#pragma unroll
  for (int g = 0; g < 4; ++g) {
    const int n = nbase + g * 32 + nc;
    const float bx = bxh[n];
#pragma unroll
    for (int j = 0; j < 8; ++j) {
      const int mg = mbase + mloc + half * 8 + j;
      float v = acc[g][j] + bx;
      if (mg < BB) v += img[mg * HH + (n & (HH - 1))];  // img4 add at t==0
      xproj[(size_t)mg * GG + n] = v;
    }
  }
}

// ---------------------------------------------------------------------------
// Kernel B (per timestep): gates = xproj[t] + h_prev @ Whh^T + bhh; LSTM update.
// M = 64 (all batch in one WG), K = 1024; each WG owns 32 hidden columns,
// each wave owns matching columns of all 4 gate blocks -> fused activation.
// grid = (HH/32), block = 256 (8 waves).
// ---------------------------------------------------------------------------
__global__ __launch_bounds__(256) void k_step(
    const bf16_t* __restrict__ h_prev, const bf16_t* __restrict__ Whh_bf,
    const float* __restrict__ bhh, const float* __restrict__ xproj_t,
    float* __restrict__ cst, bf16_t* __restrict__ h_out) {
  __shared__ bf16_t abuf[2][64 * APAD];
  const int tid = threadIdx.x, lane = tid & 31, w = tid >> 5;
  const int half = lane >> 4, nc = lane & 15;
  const int m0 = (w & 3) * 16;
  const int n0 = blockIdx.x * 32 + (w >> 2) * 16;

  v8f acc[4];
#pragma unroll
  for (int g = 0; g < 4; ++g)
#pragma unroll
    for (int j = 0; j < 8; ++j) acc[g][j] = 0.0f;

  stage64x32(&abuf[0][0], h_prev, HH, 0, tid);
  __syncthreads();
  const int KT = HH / 32;
  for (int kt = 0; kt < KT; ++kt) {
    const int buf = kt & 1;
    if (kt + 1 < KT) stage64x32(&abuf[buf ^ 1][0], h_prev, HH, (kt + 1) * 32, tid);
    v16bf a = lds_frag(&abuf[buf][0], m0, lane);
#pragma unroll
    for (int g = 0; g < 4; ++g) {
      v16bf b = g_frag(Whh_bf, HH, g * HH + n0, kt * 32, lane);  // Whh row g*H+n is contiguous over k
      acc[g] = wmma_bf16(a, b, acc[g]);
    }
    __syncthreads();
  }

  const int n = n0 + nc;
  const float bi = bhh[n], bf_ = bhh[HH + n], bg = bhh[2 * HH + n], bo = bhh[3 * HH + n];
#pragma unroll
  for (int j = 0; j < 8; ++j) {
    const int m = m0 + half * 8 + j;
    const size_t ro = (size_t)m * GG;
    const float gi = acc[0][j] + xproj_t[ro + n] + bi;
    const float gf = acc[1][j] + xproj_t[ro + HH + n] + bf_;
    const float gg = acc[2][j] + xproj_t[ro + 2 * HH + n] + bg;
    const float go = acc[3][j] + xproj_t[ro + 3 * HH + n] + bo;
    const float co = cst[m * HH + n];
    const float cn = sigm(gf) * co + sigm(gi) * tanhf(gg);
    const float hn = sigm(go) * tanhf(cn);
    cst[m * HH + n] = cn;
    h_out[m * HH + n] = (bf16_t)hn;
  }
}

// ---------------------------------------------------------------------------
// Kernel C: out = log_softmax(hs @ Wout^T + bout).
// One WG per 16 rows; WG holds full 512-wide logit rows in LDS -> fused LSE.
// grid = ((T*B)/16), block = 256 (8 waves, wave w owns cols [w*64, w*64+64)).
// ---------------------------------------------------------------------------
__global__ __launch_bounds__(256) void k_out(
    const bf16_t* __restrict__ hs, const bf16_t* __restrict__ Wout_bf,
    const float* __restrict__ bout, float* __restrict__ out) {
  __shared__ float logits[16 * 512];
  __shared__ float smax[256];
  __shared__ float ssum[256];
  const int tid = threadIdx.x, lane = tid & 31, w = tid >> 5;
  const int half = lane >> 4, nc = lane & 15;
  const int m0 = blockIdx.x * 16;

  v8f acc[4];
#pragma unroll
  for (int g = 0; g < 4; ++g)
#pragma unroll
    for (int j = 0; j < 8; ++j) acc[g][j] = 0.0f;

  const int KT = HH / 32;
  for (int kt = 0; kt < KT; ++kt) {
    v16bf a = g_frag(hs, HH, m0, kt * 32, lane);
#pragma unroll
    for (int cb = 0; cb < 4; ++cb) {
      v16bf b = g_frag(Wout_bf, HH, w * 64 + cb * 16, kt * 32, lane);
      acc[cb] = wmma_bf16(a, b, acc[cb]);
    }
  }

#pragma unroll
  for (int cb = 0; cb < 4; ++cb) {
    const int col = w * 64 + cb * 16 + nc;
    const float bo = bout[col];
#pragma unroll
    for (int j = 0; j < 8; ++j) {
      const int r = half * 8 + j;
      logits[r * 512 + col] = acc[cb][j] + bo;
    }
  }
  __syncthreads();

  // log_softmax over 512 cols per row; 16 threads per row.
  const int r = tid >> 4, c16 = tid & 15;
  float mx = -INFINITY;
  for (int c = c16; c < 512; c += 16) mx = fmaxf(mx, logits[r * 512 + c]);
  smax[tid] = mx;
  __syncthreads();
  float rowmax = -INFINITY;
#pragma unroll
  for (int i = 0; i < 16; ++i) rowmax = fmaxf(rowmax, smax[r * 16 + i]);
  float s = 0.0f;
  for (int c = c16; c < 512; c += 16) s += expf(logits[r * 512 + c] - rowmax);
  ssum[tid] = s;
  __syncthreads();
  float rowsum = 0.0f;
#pragma unroll
  for (int i = 0; i < 16; ++i) rowsum += ssum[r * 16 + i];
  const float lse = rowmax + logf(rowsum);
  for (int c = c16; c < 512; c += 16)
    out[(size_t)(m0 + r) * 512 + c] = logits[r * 512 + c] - lse;
}

// ---------------------------------------------------------------------------
extern "C" void kernel_launch(void* const* d_in, const int* in_sizes, int n_in,
                              void* d_out, int out_size, void* d_ws, size_t ws_size,
                              hipStream_t stream) {
  (void)in_sizes; (void)n_in; (void)out_size; (void)ws_size;
  const float* inp  = (const float*)d_in[0];
  const float* img  = (const float*)d_in[1];
  const float* Wxh  = (const float*)d_in[2];
  const float* bxh  = (const float*)d_in[3];
  const float* Whh  = (const float*)d_in[4];
  const float* bhh  = (const float*)d_in[5];
  const float* Wout = (const float*)d_in[6];
  const float* bout = (const float*)d_in[7];
  float* out = (float*)d_out;

  // Workspace carve-up (all sizes 256B multiples).
  char* p = (char*)d_ws;
  bf16_t* inp_bf  = (bf16_t*)p; p += (size_t)TT * BB * INS * 2;   // 16 MB
  bf16_t* Wxh_bf  = (bf16_t*)p; p += (size_t)GG * INS * 2;        //  4 MB
  bf16_t* Whh_bf  = (bf16_t*)p; p += (size_t)GG * HH * 2;         //  8 MB
  bf16_t* Wout_bf = (bf16_t*)p; p += (size_t)INS * HH * 2;        //  1 MB
  bf16_t* h0      = (bf16_t*)p; p += (size_t)BB * HH * 2;         // 128 KB
  float*  cst     = (float*)p;  p += (size_t)BB * HH * 4;         // 256 KB
  bf16_t* hsbuf   = (bf16_t*)p; p += (size_t)TT * BB * HH * 2;    // 32 MB
  float*  xproj   = (float*)p;  p += (size_t)TT * BB * GG * 4;    // 268 MB

  // 1) bf16 conversions (weights stay L2-resident afterwards).
  k_cvt_bf16<<<(TT * BB * INS + 255) / 256, 256, 0, stream>>>(inp, inp_bf, TT * BB * INS);
  k_cvt_bf16<<<(GG * INS + 255) / 256, 256, 0, stream>>>(Wxh, Wxh_bf, GG * INS);
  k_cvt_bf16<<<(GG * HH + 255) / 256, 256, 0, stream>>>(Whh, Whh_bf, GG * HH);
  k_cvt_bf16<<<(INS * HH + 255) / 256, 256, 0, stream>>>(Wout, Wout_bf, INS * HH);
  hipMemsetAsync(h0, 0, (size_t)BB * HH * 2, stream);
  hipMemsetAsync(cst, 0, (size_t)BB * HH * 4, stream);

  // 2) Input projection for all timesteps.
  k_xproj<<<dim3(GG / 128, (TT * BB) / 64), 256, 0, stream>>>(inp_bf, Wxh_bf, bxh, img, xproj);

  // 3) Sequential recurrence; h is written as bf16 directly into hs[t].
  for (int t = 0; t < TT; ++t) {
    const bf16_t* hp = (t == 0) ? h0 : (hsbuf + (size_t)(t - 1) * BB * HH);
    k_step<<<dim3(HH / 32), 256, 0, stream>>>(
        hp, Whh_bf, bhh, xproj + (size_t)t * BB * GG, cst,
        hsbuf + (size_t)t * BB * HH);
  }

  // 4) Output projection + fused log_softmax.
  k_out<<<dim3((TT * BB) / 16), 256, 0, stream>>>(hsbuf, Wout_bf, bout, out);
}
